// SwinTransformerBlock_45457933861395
// MI455X (gfx1250) — compile-verified, matrix operand order checked by reading, not run on hardware
//
#include <hip/hip_runtime.h>
#include <hip/hip_bf16.h>

// ---------------------------------------------------------------------------
// Swin dual-branch block for MI455X (gfx1250), wave32 + WMMA f16->f32.
// ---------------------------------------------------------------------------

typedef __attribute__((ext_vector_type(16))) _Float16 v16h;
typedef __attribute__((ext_vector_type(8)))  _Float16 h8;
typedef __attribute__((ext_vector_type(8)))  float    v8f;

static constexpr int Hh     = 56;
static constexpr int Ww     = 56;
static constexpr int WS     = 7;
static constexpr int SHIFT  = 3;
static constexpr int C      = 256;
static constexpr int NH     = 8;
static constexpr int HD     = 32;     // C / NH
static constexpr int Ntok   = 49;     // WS*WS
static constexpr int NP     = 64;     // padded tokens
static constexpr int NW     = 64;     // (H/WS)*(W/WS)
static constexpr int Bsz    = 16;
static constexpr int L      = Hh * Ww;          // 3136
static constexpr int R      = Bsz * NW * Ntok;  // 50176 == Bsz*L
static constexpr int CH     = 1024;   // hidden
static constexpr float SCALE = 0.17677669529663687f; // HD^-0.5
static constexpr float LN_EPS = 1e-5f;
static constexpr float NEGPAD = -10000.f;

__device__ __forceinline__ int grp(int h) { return h < (Hh - WS) ? 0 : (h < (Hh - SHIFT) ? 1 : 2); }
__device__ __forceinline__ int imin(int a, int b) { return a < b ? a : b; }

// ---------------------------------------------------------------------------
// WMMA 16x32 f16 fragment from LDS, arbitrary row stride (halfs), ISA 7.12.2:
// lanes 0-15 -> row = lane&15, K {0..7,16..23}; lanes 16-31 -> K {8..15,24..31}
// ---------------------------------------------------------------------------
__device__ __forceinline__ v16h frag_lds_s(const _Float16* base, int stride, int lane) {
  int r = lane & 15, kb = (lane & 16) ? 8 : 0;
  const _Float16* p = base + r * stride;
  h8 lo = *(const h8*)(p + kb);
  h8 hi = *(const h8*)(p + 16 + kb);
  v16h v;
#pragma unroll
  for (int t = 0; t < 8; ++t) { v[t] = lo[t]; v[8 + t] = hi[t]; }
  return v;
}

// ---------------------------------------------------------------------------
// K1: LN1 + cyclic shift (-3,-3) + window partition, f32 -> f16
// ---------------------------------------------------------------------------
__global__ __launch_bounds__(256) void k_ln_shift_part(const float* __restrict__ x,
                                                       const float* __restrict__ g,
                                                       const float* __restrict__ b,
                                                       _Float16* __restrict__ out) {
  int r = blockIdx.x, c = threadIdx.x;
  int w = r / Ntok, n = r % Ntok;
  int bi = w >> 6, wi = w & 63;
  int wh = wi >> 3, ww = wi & 7;
  int hh = wh * WS + n / WS, wp = ww * WS + n % WS;
  int src = bi * L + ((hh + SHIFT) % Hh) * Ww + ((wp + SHIFT) % Ww);
  float v = x[(size_t)src * C + c];

  __shared__ float red[256];
  red[c] = v; __syncthreads();
#pragma unroll
  for (int s = 128; s > 0; s >>= 1) { if (c < s) red[c] += red[c + s]; __syncthreads(); }
  float mean = red[0] * (1.f / 256.f);
  __syncthreads();
  float d = v - mean;
  red[c] = d * d; __syncthreads();
#pragma unroll
  for (int s = 128; s > 0; s >>= 1) { if (c < s) red[c] += red[c + s]; __syncthreads(); }
  float var = red[0] * (1.f / 256.f);
  out[(size_t)r * C + c] = (_Float16)(d * rsqrtf(var + LN_EPS) * g[c] + b[c]);
}

// ---------------------------------------------------------------------------
// K2: un-window + roll(+3,+3) + residual add + LN2 -> f16 for MLP
// ---------------------------------------------------------------------------
__global__ __launch_bounds__(256) void k_unshift_resid_ln(const float* __restrict__ xp,
                                                          const float* __restrict__ xin,
                                                          const float* __restrict__ g2,
                                                          const float* __restrict__ b2,
                                                          float* __restrict__ xres,
                                                          _Float16* __restrict__ lnout) {
  int r = blockIdx.x, c = threadIdx.x;
  int bi = r / L, l = r % L;
  int h = l / Ww, w = l % Ww;
  int hs = (h + Hh - SHIFT) % Hh, ws = (w + Ww - SHIFT) % Ww;
  int rsrc = (bi * NW + (hs / WS) * 8 + ws / WS) * Ntok + (hs % WS) * WS + (ws % WS);
  float v = xp[(size_t)rsrc * C + c] + xin[(size_t)r * C + c];
  xres[(size_t)r * C + c] = v;

  __shared__ float red[256];
  red[c] = v; __syncthreads();
#pragma unroll
  for (int s = 128; s > 0; s >>= 1) { if (c < s) red[c] += red[c + s]; __syncthreads(); }
  float mean = red[0] * (1.f / 256.f);
  __syncthreads();
  float d = v - mean;
  red[c] = d * d; __syncthreads();
#pragma unroll
  for (int s = 128; s > 0; s >>= 1) { if (c < s) red[c] += red[c + s]; __syncthreads(); }
  float var = red[0] * (1.f / 256.f);
  lnout[(size_t)r * C + c] = (_Float16)(d * rsqrtf(var + LN_EPS) * g2[c] + b2[c]);
}

// ---------------------------------------------------------------------------
// Weight transpose + f32 -> f16:  src [O][K] -> dst [K][O]
// ---------------------------------------------------------------------------
__global__ __launch_bounds__(256) void k_wtrans(const float* __restrict__ src,
                                                _Float16* __restrict__ dst, int O, int K) {
  int i = blockIdx.x * 256 + threadIdx.x;
  if (i < O * K) {
    int o = i / K, k = i % K;
    dst[(size_t)k * O + o] = (_Float16)src[i];
  }
}

// ---------------------------------------------------------------------------
// Generic WMMA GEMM: out[M][Nn] = A16[M][K] @ Bt16[K][Nn] + bias, epilogue EPI
//   EPI 0: f16 store  EPI 1: f32 store  EPI 2: gelu->f16  EPI 3: +resid->f32
// Block: 256 threads (8 waves), 64x64 tile, K step 32. M,Nn%64==0, K%32==0.
// ---------------------------------------------------------------------------
template <int EPI>
__global__ __launch_bounds__(256) void k_gemm(const _Float16* __restrict__ A,
                                              const _Float16* __restrict__ Bt,
                                              const float* __restrict__ bias,
                                              _Float16* __restrict__ outH,
                                              float* __restrict__ outF,
                                              const float* __restrict__ resid,
                                              int Nn, int K) {
  __shared__ __align__(16) _Float16 lA[64 * 32];
  __shared__ __align__(16) _Float16 lB[64 * 32]; // stored [n][k]
  int bm = blockIdx.x, bn = blockIdx.y;
  int tid = threadIdx.x, wv = tid >> 5, lane = tid & 31;
  v8f acc0 = {}; v8f acc1 = {};
  int ti0 = (wv >> 2) * 16;
  int tj  = (wv & 3) * 16;

  int arow = tid >> 2, ach = (tid & 3) * 8;
  int bk = tid >> 3, bn0 = (tid & 7) * 8;

  for (int k0 = 0; k0 < K; k0 += 32) {
    const h8* asrc = (const h8*)(A + (size_t)(bm * 64 + arow) * K + k0 + ach);
    *(h8*)(lA + arow * 32 + ach) = *asrc;
    const _Float16* bsrc = Bt + (size_t)(k0 + bk) * Nn + bn * 64 + bn0;
    h8 bv = *(const h8*)bsrc;
#pragma unroll
    for (int t = 0; t < 8; ++t) lB[(bn0 + t) * 32 + bk] = bv[t];
    if (k0 + 32 < K)
      __builtin_prefetch(A + (size_t)(bm * 64 + arow) * K + k0 + 32, 0, 1);
    __syncthreads();

    v16h bf = frag_lds_s(lB + tj * 32, 32, lane);
    v16h a0 = frag_lds_s(lA + ti0 * 32, 32, lane);
    v16h a1 = frag_lds_s(lA + (ti0 + 32) * 32, 32, lane);
    acc0 = __builtin_amdgcn_wmma_f32_16x16x32_f16(false, a0, false, bf, (short)0, acc0, false, false);
    acc1 = __builtin_amdgcn_wmma_f32_16x16x32_f16(false, a1, false, bf, (short)0, acc1, false, false);
    __syncthreads();
  }

  int gcol = bn * 64 + tj + (lane & 15);
  float bval = bias[gcol];
  int rb = bm * 64 + ti0 + ((lane & 16) ? 8 : 0);
#pragma unroll
  for (int t2 = 0; t2 < 2; ++t2) {
    v8f acc = t2 ? acc1 : acc0;
    int rbase = rb + t2 * 32;
#pragma unroll
    for (int rr = 0; rr < 8; ++rr) {
      size_t idx = (size_t)(rbase + rr) * Nn + gcol;
      float val = acc[rr] + bval;
      if (EPI == 0) {
        outH[idx] = (_Float16)val;
      } else if (EPI == 1) {
        outF[idx] = val;
      } else if (EPI == 2) {
        float ge = 0.5f * val * (1.f + erff(val * 0.70710678118654752f));
        outH[idx] = (_Float16)ge;
      } else {
        outF[idx] = val + resid[idx];
      }
    }
  }
}

// ---------------------------------------------------------------------------
// Attention: one block per window (1024 blocks), 256 threads (8 waves).
// LDS plan (~268 KB, needs CDNA5's 320KB/WGP):
//   SA16 [2][8][64][64] f16 logits (131 KB)
//   Sqk  [2][q|k][64][256] f16 staging (131 KB), reused as Vt [2][256][64]
//   rpbL [169][8] f32 (5.4 KB), FW/FB, tokp[49]
// QK^T via WMMA with bias folded into the C operand (scale folded into Q).
// rpb index decomposed additively: idx = rowoff(ih,iw,h) + coloff(jh,jw).
// ---------------------------------------------------------------------------
__global__ __launch_bounds__(256) void k_attn(const _Float16* __restrict__ qkv, // [2][R][768]
                                              const float* __restrict__ rpb_table,
                                              const float* __restrict__ fuse_w,
                                              const float* __restrict__ fuse_b,
                                              _Float16* __restrict__ xo) {      // [2][R][256]
  extern __shared__ char smem8[];
  _Float16* SA16 = (_Float16*)smem8;                 // 2*8*64*64 = 65536 halfs
  _Float16* Sqk  = SA16 + 2 * NH * NP * NP;          // 2*2*64*256 = 65536 halfs
  _Float16* Vt   = Sqk;                              // reused after Phase A
  float* rpbL    = (float*)(Sqk + 2 * 2 * NP * C);   // 1352 floats
  float* FW      = rpbL + 169 * NH;                  // 64
  float* FB      = FW + 64;                          // 8
  int*   tokp    = (int*)(FB + 8);                   // 49

  const size_t RQ = (size_t)R * 768;
  int w = blockIdx.x;
  int tid = threadIdx.x, wv = tid >> 5, lane = tid & 31;
  int l15 = lane & 15, msel = (lane & 16) ? 8 : 0;
  int wi = w & 63, whd = wi >> 3, wwd = wi & 7;

  // ---- stage small tables ----
  for (int i = tid; i < 169 * NH; i += 256) rpbL[i] = rpb_table[i];
  if (tid < 64) FW[tid] = fuse_w[tid];
  if (tid < 8)  FB[tid] = fuse_b[tid];
  if (tid < Ntok) {
    int ih = tid / WS, iw = tid % WS;
    int gg = grp(whd * WS + ih) * 3 + grp(wwd * WS + iw);
    tokp[tid] = (ih << 8) | (iw << 4) | gg;
  }

  // ---- stage Q (pre-scaled) and K, zero-padded to 64 rows ----
  for (int ch = tid; ch < 2 * 2 * NP * 32; ch += 256) {
    int c8  = ch & 31;
    int row = (ch >> 5) & 63;
    int mat = (ch >> 11) & 1;
    int br  = (ch >> 12) & 1;
    h8 val = {};
    if (row < Ntok) {
      val = *(const h8*)(qkv + (size_t)br * RQ + ((size_t)w * Ntok + row) * 768 + mat * C + c8 * 8);
      if (mat == 0) {
#pragma unroll
        for (int t = 0; t < 8; ++t) val[t] = (_Float16)((float)val[t] * SCALE);
      }
    }
    *(h8*)(Sqk + ((((br * 2 + mat) * NP + row)) << 8) + c8 * 8) = val;
  }
  __syncthreads();

  // ---- Phase A: logits = Qs K^T + (rpb + mask) via WMMA C operand ----
#pragma unroll
  for (int cc = 0; cc < 2; ++cc) {
    int c = wv * 2 + cc, br = c & 1, hh = c >> 1;
    const _Float16* Sq = Sqk + ((size_t)(br * 2 + 0) * NP << 8) + hh * HD;
    const _Float16* Sk = Sqk + ((size_t)(br * 2 + 1) * NP << 8) + hh * HD;
    _Float16* SAc = SA16 + ((size_t)(br * NH + hh) * NP << 6);
    for (int ti = 0; ti < 4; ++ti) {
      v16h a = frag_lds_s(Sq + (ti * 16 << 8), 256, lane);
      int rb = ti * 16 + msel;
      // hoisted per-row descriptors (independent of tj)
      int rowoff[8], gi8[8];
      bool rok[8];
#pragma unroll
      for (int rr = 0; rr < 8; ++rr) {
        int row = rb + rr;
        int tpi = tokp[imin(row, Ntok - 1)];
        int ih = tpi >> 8, iw = (tpi >> 4) & 15;
        gi8[rr]    = tpi & 15;
        rowoff[rr] = (ih * 13 + iw) * NH + hh;
        rok[rr]    = row < Ntok;
      }
      for (int tj = 0; tj < 4; ++tj) {
        v16h bf = frag_lds_s(Sk + (tj * 16 << 8), 256, lane);
        int col = tj * 16 + l15;
        int tpj = tokp[imin(col, Ntok - 1)];
        int jh = tpj >> 8, jw = (tpj >> 4) & 15, gj = tpj & 15;
        int coloff = ((6 - jh) * 13 + (6 - jw)) * NH;
        bool colok = col < Ntok;
        v8f cb;
#pragma unroll
        for (int rr = 0; rr < 8; ++rr) {
          float rp = rpbL[rowoff[rr] + coloff];
          float mk = (gi8[rr] != gj) ? -100.f : 0.f;
          cb[rr] = (colok && rok[rr]) ? (rp + mk) : NEGPAD;
        }
        v8f d = __builtin_amdgcn_wmma_f32_16x16x32_f16(false, a, false, bf, (short)0, cb, false, false);
#pragma unroll
        for (int rr = 0; rr < 8; ++rr)
          SAc[((rb + rr) << 6) + col] = (_Float16)d[rr];
      }
    }
  }
  __syncthreads();

  // ---- stage V transposed into Vt [br][256][64] (overwrites Sqk) ----
  for (int ch = tid; ch < 2 * NP * 32; ch += 256) {
    int c8  = ch & 31;
    int row = (ch >> 5) & 63;
    int br  = (ch >> 11) & 1;
    h8 val = {};
    if (row < Ntok)
      val = *(const h8*)(qkv + (size_t)br * RQ + ((size_t)w * Ntok + row) * 768 + 2 * C + c8 * 8);
    _Float16* dst = Vt + ((size_t)br * C << 6);
#pragma unroll
    for (int t = 0; t < 8; ++t) dst[((c8 * 8 + t) << 6) + row] = val[t];
  }

  // ---- Phase B1: cross-branch head-mix sigmoid gate (overwrite branch 0) ----
  for (int e = tid; e < Ntok * Ntok; e += 256) {
    int i = e / Ntok, j = e % Ntok;
    float a0[NH], df[NH];
#pragma unroll
    for (int hh = 0; hh < NH; ++hh) {
      float v0 = (float)SA16[(((size_t)(0 * NH + hh) * NP + i) << 6) + j];
      float v1 = (float)SA16[(((size_t)(1 * NH + hh) * NP + i) << 6) + j];
      a0[hh] = v0; df[hh] = v1 - v0;
    }
#pragma unroll
    for (int o = 0; o < NH; ++o) {
      float s = FB[o];
#pragma unroll
      for (int hh = 0; hh < NH; ++hh) s += FW[o * NH + hh] * df[hh];
      float gate = 1.f / (1.f + expf(-s));
      SA16[(((size_t)(0 * NH + o) * NP + i) << 6) + j] = (_Float16)(a0[o] + df[o] * gate);
    }
  }
  __syncthreads();

  // ---- Phase B2: row softmax over 64 cols (padded cols -> ~0), 784 rows ----
  for (int rr = tid; rr < 2 * NH * Ntok; rr += 256) {
    int brh = rr / Ntok, i = rr % Ntok;
    _Float16* row = SA16 + (((size_t)brh * NP + i) << 6);
    float mx = -1e30f;
    for (int j = 0; j < NP; ++j) mx = fmaxf(mx, (float)row[j]);
    float s = 0.f;
    for (int j = 0; j < NP; ++j) { float e = expf((float)row[j] - mx); row[j] = (_Float16)e; s += e; }
    float inv = 1.f / s;
    for (int j = 0; j < NP; ++j) row[j] = (_Float16)((float)row[j] * inv);
  }
  __syncthreads();

  // ---- Phase C: out = A @ V ----
#pragma unroll
  for (int cc = 0; cc < 2; ++cc) {
    int c = wv * 2 + cc, br = c & 1, hh = c >> 1;
    const _Float16* Ab = SA16 + ((size_t)(br * NH + hh) * NP << 6);
    const _Float16* Vb = Vt + ((size_t)(br * C + hh * HD) << 6);
    for (int ti = 0; ti < 4; ++ti) {
      for (int tn = 0; tn < 2; ++tn) {
        v8f d = {};
#pragma unroll
        for (int ks = 0; ks < 2; ++ks) {
          v16h a  = frag_lds_s(Ab + (ti * 16 << 6) + ks * 32, NP, lane);
          v16h bf = frag_lds_s(Vb + (tn * 16 << 6) + ks * 32, NP, lane);
          d = __builtin_amdgcn_wmma_f32_16x16x32_f16(false, a, false, bf, (short)0, d, false, false);
        }
        int coln = tn * 16 + l15;
        int rb = ti * 16 + msel;
        _Float16* outp = xo + (size_t)br * R * C + (size_t)w * Ntok * C + hh * HD + coln;
#pragma unroll
        for (int rr = 0; rr < 8; ++rr) {
          int row = rb + rr;
          if (row < Ntok) outp[(size_t)row * C] = (_Float16)d[rr];
        }
      }
    }
  }
}

// ---------------------------------------------------------------------------
// Host-side launch
// ---------------------------------------------------------------------------
extern "C" void kernel_launch(void* const* d_in, const int* in_sizes, int n_in,
                              void* d_out, int out_size, void* d_ws, size_t ws_size,
                              hipStream_t stream) {
  (void)in_sizes; (void)n_in; (void)out_size; (void)ws_size;

  const float* x_in[2]   = { (const float*)d_in[0],  (const float*)d_in[1]  };
  const float* ln1g[2]   = { (const float*)d_in[2],  (const float*)d_in[4]  };
  const float* ln1b[2]   = { (const float*)d_in[3],  (const float*)d_in[5]  };
  const float* qkvw[2]   = { (const float*)d_in[6],  (const float*)d_in[8]  };
  const float* qkvb[2]   = { (const float*)d_in[7],  (const float*)d_in[9]  };
  const float* rpb       =   (const float*)d_in[10];
  const float* fusew     =   (const float*)d_in[11];
  const float* fuseb     =   (const float*)d_in[12];
  const float* projw[2]  = { (const float*)d_in[13], (const float*)d_in[15] };
  const float* projb[2]  = { (const float*)d_in[14], (const float*)d_in[16] };
  const float* ln2g[2]   = { (const float*)d_in[17], (const float*)d_in[19] };
  const float* ln2b[2]   = { (const float*)d_in[18], (const float*)d_in[20] };
  const float* fc1w[2]   = { (const float*)d_in[21], (const float*)d_in[25] };
  const float* fc1b[2]   = { (const float*)d_in[22], (const float*)d_in[26] };
  const float* fc2w[2]   = { (const float*)d_in[23], (const float*)d_in[27] };
  const float* fc2b[2]   = { (const float*)d_in[24], (const float*)d_in[28] };
  float* outp = (float*)d_out;

  // workspace carve-up
  char* p = (char*)d_ws;
  auto alloc = [&](size_t bytes) { char* r = p; p += (bytes + 255) & ~(size_t)255; return r; };
  const size_t Rz = (size_t)R;
  _Float16* xw16   = (_Float16*)alloc(2 * Rz * C * 2);
  _Float16* qkv16  = (_Float16*)alloc(2 * Rz * 3 * C * 2);
  _Float16* xo16   = (_Float16*)alloc(2 * Rz * C * 2);
  float*    xp32   = (float*)   alloc(2 * Rz * C * 4);
  float*    xres   = (float*)   alloc(2 * Rz * C * 4);
  _Float16* ln2h   = (_Float16*)alloc(2 * Rz * C * 2);
  _Float16* h16    = (_Float16*)alloc(2 * Rz * CH * 2);
  _Float16* wq16   = (_Float16*)alloc(2 * (size_t)3 * C * C * 2);
  _Float16* wp16   = (_Float16*)alloc(2 * (size_t)C * C * 2);
  _Float16* w116   = (_Float16*)alloc(2 * (size_t)CH * C * 2);
  _Float16* w216   = (_Float16*)alloc(2 * (size_t)C * CH * 2);

  // ---- weight transpose/convert ----
  for (int br = 0; br < 2; ++br) {
    k_wtrans<<<(3 * C * C + 255) / 256, 256, 0, stream>>>(qkvw[br], wq16 + (size_t)br * 3 * C * C, 3 * C, C);
    k_wtrans<<<(C * C + 255) / 256,     256, 0, stream>>>(projw[br], wp16 + (size_t)br * C * C, C, C);
    k_wtrans<<<(CH * C + 255) / 256,    256, 0, stream>>>(fc1w[br], w116 + (size_t)br * CH * C, CH, C);
    k_wtrans<<<(C * CH + 255) / 256,    256, 0, stream>>>(fc2w[br], w216 + (size_t)br * C * CH, C, CH);
  }

  // ---- LN1 + shift + window partition ----
  for (int br = 0; br < 2; ++br)
    k_ln_shift_part<<<R, 256, 0, stream>>>(x_in[br], ln1g[br], ln1b[br], xw16 + (size_t)br * Rz * C);

  // ---- QKV GEMM: [R,256] @ [256,768] ----
  for (int br = 0; br < 2; ++br)
    k_gemm<0><<<dim3(R / 64, (3 * C) / 64), 256, 0, stream>>>(
        xw16 + (size_t)br * Rz * C, wq16 + (size_t)br * 3 * C * C, qkvb[br],
        qkv16 + (size_t)br * Rz * 3 * C, nullptr, nullptr, 3 * C, C);

  // ---- attention (both branches fused, one window per block) ----
  const size_t attn_smem = (size_t)(2 * NH * NP * NP + 2 * 2 * NP * C) * sizeof(_Float16)
                         + (size_t)(169 * NH + 64 + 8) * sizeof(float)
                         + (size_t)Ntok * sizeof(int);
  k_attn<<<Bsz * NW, 256, attn_smem, stream>>>(qkv16, rpb, fusew, fuseb, xo16);

  // ---- proj GEMM: [R,256] @ [256,256] -> f32 ----
  for (int br = 0; br < 2; ++br)
    k_gemm<1><<<dim3(R / 64, C / 64), 256, 0, stream>>>(
        xo16 + (size_t)br * Rz * C, wp16 + (size_t)br * C * C, projb[br],
        nullptr, xp32 + (size_t)br * Rz * C, nullptr, C, C);

  // ---- un-shift + residual + LN2 ----
  for (int br = 0; br < 2; ++br)
    k_unshift_resid_ln<<<R, 256, 0, stream>>>(
        xp32 + (size_t)br * Rz * C, x_in[br], ln2g[br], ln2b[br],
        xres + (size_t)br * Rz * C, ln2h + (size_t)br * Rz * C);

  // ---- fc1 GEMM + GELU: [R,256] @ [256,1024] ----
  for (int br = 0; br < 2; ++br)
    k_gemm<2><<<dim3(R / 64, CH / 64), 256, 0, stream>>>(
        ln2h + (size_t)br * Rz * C, w116 + (size_t)br * CH * C, fc1b[br],
        h16 + (size_t)br * Rz * CH, nullptr, nullptr, CH, C);

  // ---- fc2 GEMM + bias + residual -> d_out ----
  for (int br = 0; br < 2; ++br)
    k_gemm<3><<<dim3(R / 64, C / 64), 256, 0, stream>>>(
        h16 + (size_t)br * Rz * CH, w216 + (size_t)br * C * CH, fc2b[br],
        nullptr, outp + (size_t)br * Rz * C, xres + (size_t)br * Rz * C, C, CH);
}
